// MultiHeadAttention_59536836657243
// MI455X (gfx1250) — matrix operand order repaired and checked by reading, not compiled, not run on hardware
//
#include <hip/hip_runtime.h>
#include <hip/hip_bf16.h>

// ---------------------------------------------------------------------------
// MultiHeadAttention forward for gfx1250 (MI455X), bf16 WMMA pipeline with
// async global->LDS staging (ASYNCcnt) and double-buffered LDS tiles.
// B=4, S=2048, D=1024, H=16, HD=64
// ---------------------------------------------------------------------------

#define B_SZ 4
#define S_LEN 2048
#define D_SZ 1024
#define HEADS 16
#define HDIM 64
#define QKV_W (3 * D_SZ)   // 3072

typedef __attribute__((ext_vector_type(16))) __bf16 v16bf;
typedef __attribute__((ext_vector_type(8)))  __bf16 v8bf;
typedef __attribute__((ext_vector_type(8)))  float  v8f;
typedef __attribute__((ext_vector_type(4)))  int    v4i;

typedef __attribute__((address_space(1))) v4i* gv4i_ptr;
typedef __attribute__((address_space(3))) v4i* lv4i_ptr;

#if defined(__has_builtin)
#if __has_builtin(__builtin_amdgcn_global_load_async_to_lds_b128) && \
    __has_builtin(__builtin_amdgcn_s_wait_asynccnt)
#define HAVE_ASYNC_LDS 1
#endif
#endif
#ifndef HAVE_ASYNC_LDS
#define HAVE_ASYNC_LDS 0
#endif

// 16-byte global -> LDS copy. Async path uses GLOBAL_LOAD_ASYNC_TO_LDS_B128
// (tracked by ASYNCcnt, no VGPR round trip).
__device__ __forceinline__ void copy16B_g2l(__bf16* l, const __bf16* g) {
#if HAVE_ASYNC_LDS
  __builtin_amdgcn_global_load_async_to_lds_b128(
      (gv4i_ptr)(void*)(g),
      (lv4i_ptr)(void*)(l),
      /*offset=*/0, /*cpol=*/0);
#else
  *(v8bf*)l = *(const v8bf*)g;
#endif
}

__device__ __forceinline__ void async_fence() {
#if HAVE_ASYNC_LDS
  __builtin_amdgcn_s_wait_asynccnt(0);
#endif
}

// Build a 16-element bf16 A/B fragment from two contiguous 8-element chunks.
__device__ __forceinline__ v16bf load_frag2(const __bf16* p0, const __bf16* p1) {
  v8bf lo = *(const v8bf*)p0;
  v8bf hi = *(const v8bf*)p1;
  v16bf out;
#pragma unroll
  for (int i = 0; i < 8; ++i) { out[i] = lo[i]; out[i + 8] = hi[i]; }
  return out;
}

__device__ __forceinline__ v8f wmma_bf16(v16bf a, v16bf b, v8f c) {
  return __builtin_amdgcn_wmma_f32_16x16x32_bf16(
      /*neg_a=*/false, a, /*neg_b=*/false, b,
      /*c_mod=*/(short)0, c, /*reuse_a=*/false, /*reuse_b=*/false);
}

// ---------------------------------------------------------------------------
// fp32 -> bf16 conversion (one-time for x, W_qkv, W_out)
// ---------------------------------------------------------------------------
__global__ __launch_bounds__(256) void f32_to_bf16_kernel(
    const float* __restrict__ in, __bf16* __restrict__ out, int n) {
  int i = blockIdx.x * 256 + threadIdx.x;
  if (i < n) out[i] = (__bf16)in[i];
}

// ---------------------------------------------------------------------------
// bf16 GEMM: C[M,N] = A[M,K] * Bw[N,K]^T + bias[N]
//   block tile 128x128, 8 waves (wave32), per-wave 32x64 (2x4 WMMA accums)
//   double-buffered LDS, async staging. Output bf16 (Cb) or fp32 (Cf).
// ---------------------------------------------------------------------------
__global__ __launch_bounds__(256) void gemm_bf16_kernel(
    const __bf16* __restrict__ A, const __bf16* __restrict__ Bw,
    const float* __restrict__ bias, float* __restrict__ Cf,
    __bf16* __restrict__ Cb, int M, int N, int K) {
  __shared__ __bf16 As[2][128][32];
  __shared__ __bf16 Bs[2][128][32];

  const int tid  = threadIdx.x;
  const int wid  = tid >> 5;
  const int lane = tid & 31;
  const int hl   = lane >> 4;   // lane half (0/1)
  const int r    = lane & 15;   // lane within half
  const int wm   = wid & 3;     // 4 waves along M
  const int wn   = wid >> 2;    // 2 waves along N
  const int mBase = wm * 32;
  const int nBase = wn * 64;
  const int blockM = blockIdx.y * 128;
  const int blockN = blockIdx.x * 128;

  const int sRow = tid >> 1;          // 0..127
  const int sCol = (tid & 1) * 16;    // 0 or 16

  auto stage = [&](int buf, int kk) {
    const __bf16* ga = A + (size_t)(blockM + sRow) * K + kk + sCol;
    const __bf16* gb = Bw + (size_t)(blockN + sRow) * K + kk + sCol;
#if HAVE_ASYNC_LDS
    copy16B_g2l(&As[buf][sRow][sCol],     ga);
    copy16B_g2l(&As[buf][sRow][sCol + 8], ga + 8);
    copy16B_g2l(&Bs[buf][sRow][sCol],     gb);
    copy16B_g2l(&Bs[buf][sRow][sCol + 8], gb + 8);
#else
    v8bf a0 = *(const v8bf*)ga;
    v8bf a1 = *(const v8bf*)(ga + 8);
    v8bf b0 = *(const v8bf*)gb;
    v8bf b1 = *(const v8bf*)(gb + 8);
    *(v8bf*)&As[buf][sRow][sCol]     = a0;
    *(v8bf*)&As[buf][sRow][sCol + 8] = a1;
    *(v8bf*)&Bs[buf][sRow][sCol]     = b0;
    *(v8bf*)&Bs[buf][sRow][sCol + 8] = b1;
#endif
    if (kk + 32 < K) {
      __builtin_prefetch(ga + 32, 0, 3);
      __builtin_prefetch(gb + 32, 0, 3);
    }
  };

  v8f acc[2][4];
#pragma unroll
  for (int ms = 0; ms < 2; ++ms)
#pragma unroll
    for (int ns = 0; ns < 4; ++ns) acc[ms][ns] = (v8f)0.0f;

  stage(0, 0);
  int buf = 0;
  for (int kk = 0; kk < K; kk += 32) {
    async_fence();          // my async fills of As/Bs[buf] are done
    __syncthreads();        // everyone's fills are done, prior reads retired
    if (kk + 32 < K) stage(buf ^ 1, kk + 32);   // overlap next fill w/ compute

    v16bf af[2], bfr[4];
#pragma unroll
    for (int ms = 0; ms < 2; ++ms) {
      int row = mBase + ms * 16 + r;
      af[ms] = load_frag2(&As[buf][row][hl * 8], &As[buf][row][16 + hl * 8]);
    }
#pragma unroll
    for (int ns = 0; ns < 4; ++ns) {
      int n = nBase + ns * 16 + r;
      bfr[ns] = load_frag2(&Bs[buf][n][hl * 8], &Bs[buf][n][16 + hl * 8]);
    }
#pragma unroll
    for (int ms = 0; ms < 2; ++ms)
#pragma unroll
      for (int ns = 0; ns < 4; ++ns)
        acc[ms][ns] = wmma_bf16(af[ms], bfr[ns], acc[ms][ns]);
    buf ^= 1;
  }

  // Epilogue: bias add, store (C layout: lanes 0-15 row v, lanes 16-31 row 8+v)
  if (Cb) {
#pragma unroll
    for (int ms = 0; ms < 2; ++ms)
#pragma unroll
      for (int ns = 0; ns < 4; ++ns) {
        int col = blockN + nBase + ns * 16 + r;
        float bv = bias[col];
#pragma unroll
        for (int v = 0; v < 8; ++v) {
          int row = blockM + mBase + ms * 16 + hl * 8 + v;
          Cb[(size_t)row * N + col] = (__bf16)(acc[ms][ns][v] + bv);
        }
      }
  } else {
#pragma unroll
    for (int ms = 0; ms < 2; ++ms)
#pragma unroll
      for (int ns = 0; ns < 4; ++ns) {
        int col = blockN + nBase + ns * 16 + r;
        float bv = bias[col];
#pragma unroll
        for (int v = 0; v < 8; ++v) {
          int row = blockM + mBase + ms * 16 + hl * 8 + v;
          Cf[(size_t)row * N + col] = acc[ms][ns][v] + bv;
        }
      }
  }
}

// ---------------------------------------------------------------------------
// Flash-style attention. qkv layout per row: [H][192] = [H][q(64)|k(64)|v(64)]
// grid: (B*H, S/128); block 256 = 8 waves; wave owns 16 q rows.
// Double-buffered K (async staged) / V (register-transposed) blocks.
// ---------------------------------------------------------------------------
__global__ __launch_bounds__(256) void attn_kernel(
    const __bf16* __restrict__ qkv, __bf16* __restrict__ vals) {
  __shared__ __bf16 Kblk[2][32][64];   // 32 keys x 64 hd (row-major)
  __shared__ __bf16 Vt[2][64][40];     // transposed V: hd x keys (padded)
  __shared__ __bf16 Ps[8][16][32];     // per-wave P re-layout scratch

  const int bh = blockIdx.x;
  const int b  = bh / HEADS;
  const int h  = bh % HEADS;
  const int tid  = threadIdx.x;
  const int wid  = tid >> 5;
  const int lane = tid & 31;
  const int hl   = lane >> 4;
  const int r    = lane & 15;
  const int qRow = blockIdx.y * 128 + wid * 16;
  const size_t baseBS = (size_t)b * S_LEN;

  auto stage = [&](int buf, int j) {
    int key = tid & 31;
    int hc  = (tid >> 5) * 8;     // hd chunk 0..56
    const __bf16* srck = qkv + (baseBS + j + key) * QKV_W + h * 192 + 64 + hc;
    copy16B_g2l(&Kblk[buf][key][hc], srck);     // async K staging
    const __bf16* srcv = qkv + (baseBS + j + key) * QKV_W + h * 192 + 128 + hc;
    v8bf vv = *(const v8bf*)srcv;               // V transposed via registers
#pragma unroll
    for (int i = 0; i < 8; ++i) Vt[buf][hc + i][key] = vv[i];
  };

  // Q fragments for this wave's 16 rows (A-layout, K = HD in two 32-chunks)
  v16bf qf[2];
  {
    const __bf16* qrow = qkv + (baseBS + qRow + r) * QKV_W + h * 192;
#pragma unroll
    for (int ks = 0; ks < 2; ++ks)
      qf[ks] = load_frag2(qrow + ks * 32 + hl * 8,
                          qrow + ks * 32 + 16 + hl * 8);
  }

  v8f O[4];
#pragma unroll
  for (int ns = 0; ns < 4; ++ns) O[ns] = (v8f)0.0f;
  float mrow[8], lrow[8];
#pragma unroll
  for (int v = 0; v < 8; ++v) { mrow[v] = -3.0e38f; lrow[v] = 0.0f; }

  stage(0, 0);
  int buf = 0;
  for (int j = 0; j < S_LEN; j += 32) {
    async_fence();
    __syncthreads();
    if (j + 32 < S_LEN) stage(buf ^ 1, j + 32);

    // scores: two 16x16 tiles (keys j..j+15 and j+16..j+31), K-dim = HD(64)
    v8f s0 = (v8f)0.0f, s1 = (v8f)0.0f;
#pragma unroll
    for (int ks = 0; ks < 2; ++ks) {
      v16bf bk0 = load_frag2(&Kblk[buf][r][ks * 32 + hl * 8],
                             &Kblk[buf][r][ks * 32 + 16 + hl * 8]);
      v16bf bk1 = load_frag2(&Kblk[buf][16 + r][ks * 32 + hl * 8],
                             &Kblk[buf][16 + r][ks * 32 + 16 + hl * 8]);
      s0 = wmma_bf16(qf[ks], bk0, s0);
      s1 = wmma_bf16(qf[ks], bk1, s1);
    }

    // Online softmax update (rows striped across 16-lane halves).
    float alpha[8];
#pragma unroll
    for (int v = 0; v < 8; ++v) {
      float a0 = s0[v] * 0.125f;     // 1/sqrt(64)
      float a1 = s1[v] * 0.125f;
      float mx = fmaxf(a0, a1);
      mx = fmaxf(mx, __shfl_xor(mx, 1, 16));
      mx = fmaxf(mx, __shfl_xor(mx, 2, 16));
      mx = fmaxf(mx, __shfl_xor(mx, 4, 16));
      mx = fmaxf(mx, __shfl_xor(mx, 8, 16));
      float mnew = fmaxf(mrow[v], mx);
      float al = __expf(mrow[v] - mnew);
      float p0 = __expf(a0 - mnew);
      float p1 = __expf(a1 - mnew);
      float sm = p0 + p1;
      sm += __shfl_xor(sm, 1, 16);
      sm += __shfl_xor(sm, 2, 16);
      sm += __shfl_xor(sm, 4, 16);
      sm += __shfl_xor(sm, 8, 16);
      lrow[v] = lrow[v] * al + sm;
      mrow[v] = mnew;
      alpha[v] = al;
      // C-tile -> LDS (row = hl*8+v, col = r / 16+r)
      Ps[wid][hl * 8 + v][r]      = (__bf16)p0;
      Ps[wid][hl * 8 + v][16 + r] = (__bf16)p1;
    }
#pragma unroll
    for (int ns = 0; ns < 4; ++ns)
#pragma unroll
      for (int v = 0; v < 8; ++v) O[ns][v] *= alpha[v];

    // P back as A-fragment (16 x 32 keys), then O += P @ V
    v16bf pa = load_frag2(&Ps[wid][r][hl * 8], &Ps[wid][r][16 + hl * 8]);
#pragma unroll
    for (int ns = 0; ns < 4; ++ns) {
      v16bf bv = load_frag2(&Vt[buf][ns * 16 + r][hl * 8],
                            &Vt[buf][ns * 16 + r][16 + hl * 8]);
      O[ns] = wmma_bf16(pa, bv, O[ns]);
    }
    buf ^= 1;
  }

  // Normalize and write vals[b, s, h*64 + d] as bf16 for the output GEMM.
#pragma unroll
  for (int v = 0; v < 8; ++v) {
    float inv = 1.0f / lrow[v];
    int rowg = qRow + hl * 8 + v;
    __bf16* dst = vals + (baseBS + rowg) * D_SZ + h * HDIM;
#pragma unroll
    for (int ns = 0; ns < 4; ++ns)
      dst[ns * 16 + r] = (__bf16)(O[ns][v] * inv);
  }
}

// ---------------------------------------------------------------------------
// Host-side orchestration
// ---------------------------------------------------------------------------
extern "C" void kernel_launch(void* const* d_in, const int* in_sizes, int n_in,
                              void* d_out, int out_size, void* d_ws,
                              size_t ws_size, hipStream_t stream) {
  const float* x     = (const float*)d_in[0];   // [B,S,D]
  const float* W_qkv = (const float*)d_in[1];   // [3D,D]
  const float* b_qkv = (const float*)d_in[2];   // [3D]
  const float* W_out = (const float*)d_in[3];   // [D,D]
  const float* b_out = (const float*)d_in[4];   // [D]
  float* out = (float*)d_out;                   // [B,S,D]

  const int M = B_SZ * S_LEN;                   // 8192
  const size_t nX    = (size_t)M * D_SZ;        // 8388608
  const size_t nWqkv = (size_t)QKV_W * D_SZ;    // 3145728
  const size_t nWout = (size_t)D_SZ * D_SZ;     // 1048576
  const size_t nQKV  = (size_t)M * QKV_W;       // 25165824
  const size_t nVals = (size_t)M * D_SZ;

  char* ws = (char*)d_ws;
  size_t off = 0;
  __bf16* xb    = (__bf16*)(ws + off); off += nX * 2;
  __bf16* wqkvb = (__bf16*)(ws + off); off += nWqkv * 2;
  __bf16* woutb = (__bf16*)(ws + off); off += nWout * 2;
  __bf16* qkvb  = (__bf16*)(ws + off); off += nQKV * 2;
  __bf16* valsb = (__bf16*)(ws + off); off += nVals * 2;
  (void)off; (void)ws_size; (void)in_sizes; (void)n_in; (void)out_size;

  // 1) fp32 -> bf16 conversions
  f32_to_bf16_kernel<<<(int)((nX + 255) / 256), 256, 0, stream>>>(x, xb, (int)nX);
  f32_to_bf16_kernel<<<(int)((nWqkv + 255) / 256), 256, 0, stream>>>(W_qkv, wqkvb, (int)nWqkv);
  f32_to_bf16_kernel<<<(int)((nWout + 255) / 256), 256, 0, stream>>>(W_out, woutb, (int)nWout);

  // 2) QKV projection: qkv[M, 3D] = x @ W_qkv^T + b_qkv  (bf16 out)
  gemm_bf16_kernel<<<dim3(QKV_W / 128, M / 128), 256, 0, stream>>>(
      xb, wqkvb, b_qkv, nullptr, qkvb, M, QKV_W, D_SZ);

  // 3) Attention -> vals[M, D] (bf16)
  attn_kernel<<<dim3(B_SZ * HEADS, S_LEN / 128), 256, 0, stream>>>(qkvb, valsb);

  // 4) Output projection: out = vals @ W_out^T + b_out (fp32 out)
  gemm_bf16_kernel<<<dim3(D_SZ / 128, M / 128), 256, 0, stream>>>(
      valsb, woutb, b_out, out, nullptr, M, D_SZ, D_SZ);
}
